// ConvInteration_5007931867797
// MI455X (gfx1250) — matrix operand order repaired and checked by reading, not compiled, not run on hardware
//
#include <hip/hip_runtime.h>
#include <hip/hip_bf16.h>

// ---------------------------------------------------------------------------
// ConvInteration fused pipeline for MI455X (gfx1250, wave32, WMMA bf16).
//   K1: Ux = relu(x @ Wu) fused with per-chunk top-4 over S   (68.7 GFLOP)
//   K2: merge chunk top-4 lists -> z                           (tiny)
//   K3: out = relu((z*P) @ Q + Bb)                             (8.6 GFLOP)
// f32 -> bf16 via v_perm_b32 packing, f32 accumulate WMMA, LDS double-buffer,
// 16B-aligned LDS rows so fragments move as ds_load_b128/ds_store_b128.
// ---------------------------------------------------------------------------

typedef __attribute__((ext_vector_type(16))) __bf16 v16bf;
typedef __attribute__((ext_vector_type(8)))  float  v8f;

#define BSZ    32
#define SEQ    4096
#define FDIM   512
#define KTOP   4
#define MT     128          // block tile M
#define NT     128          // block tile N
#define KT     32           // K slice per WMMA pass
#define NSLICE (FDIM / KT)  // 16
#define SA     40           // LDS row stride in bf16 elems: 80 B, 16B-aligned
#define SB     40
#define NCHUNK (SEQ / MT)   // 32 sequence chunks

union Frag { uint4 q[2]; unsigned int u[8]; v16bf v; };

// Pack two f32 -> two bf16 (round-half-up) in one v_perm_b32.
static __device__ __forceinline__ unsigned int pack_bf2(float lo, float hi) {
    union { float f; unsigned int u; } a, b; a.f = lo; b.f = hi;
    return __builtin_amdgcn_perm(b.u + 0x8000u, a.u + 0x8000u, 0x07060302u);
}

// Branchless sorted-descending top-4 insert: 7 min/max ops, no divergence.
static __device__ __forceinline__ void ins4(float v, float t[4]) {
    float m0 = fminf(t[0], v);  t[0] = fmaxf(t[0], v);
    float m1 = fminf(t[1], m0); t[1] = fmaxf(t[1], m0);
    float m2 = fminf(t[2], m1); t[2] = fmaxf(t[2], m1);
    t[3] = fmaxf(t[3], m2);
}

// A fragment (16x32 bf16, ISA 16-bit A layout): two 16B LDS loads.
static __device__ __forceinline__ void load_af(const unsigned short* lds,
                                               int row_base, int lane,
                                               Frag& f) {
    const int M = lane & 15, half = lane >> 4;
    const uint4* rp = (const uint4*)(lds + (row_base + M) * SA);
    f.q[0] = rp[half];       // K = half*8 + 0..7   (pairs)
    f.q[1] = rp[2 + half];   // K = 16 + half*8 + 0..7
}

// B fragment (32x16 bf16): LDS tile transposed [N][K]; two 16B loads.
static __device__ __forceinline__ void load_bf(const unsigned short* lds,
                                               int col_base, int lane,
                                               Frag& f) {
    const int N = lane & 15, half = lane >> 4;
    const uint4* rp = (const uint4*)(lds + (col_base + N) * SB);
    f.q[0] = rp[half];           // K = half*8 + 0..7
    f.q[1] = rp[2 + half];       // K = 16 + half*8 + 0..7
}

// A stage: thread owns one M row, 16 contiguous K. Commit = 2 b128 stores.
static __device__ __forceinline__ void commitA(unsigned short* ldsA, int t,
                                               const float4 ra[4], float scale) {
    const int row = t >> 1, hh = (t & 1) * 16;
    uint4* dst = (uint4*)(ldsA + row * SA + hh);
    uint4 w0, w1;
    w0.x = pack_bf2(ra[0].x * scale, ra[0].y * scale);
    w0.y = pack_bf2(ra[0].z * scale, ra[0].w * scale);
    w0.z = pack_bf2(ra[1].x * scale, ra[1].y * scale);
    w0.w = pack_bf2(ra[1].z * scale, ra[1].w * scale);
    w1.x = pack_bf2(ra[2].x * scale, ra[2].y * scale);
    w1.y = pack_bf2(ra[2].z * scale, ra[2].w * scale);
    w1.z = pack_bf2(ra[3].x * scale, ra[3].y * scale);
    w1.w = pack_bf2(ra[3].z * scale, ra[3].w * scale);
    dst[0] = w0; dst[1] = w1;
}

// B stage: thread owns K rows {kp, kp+1} x 8 columns.  rb[0..1] = row kp,
// rb[2..3] = row kp+1.  Commit packs (k,k+1) pairs -> 8 b32 stores.
static __device__ __forceinline__ void commitB(unsigned short* ldsB, int t,
                                               const float4 rb[4]) {
    const int kp = (t >> 4) * 2, c = (t & 15) * 8;
    unsigned int* base = (unsigned int*)ldsB;
    const float r0[8] = {rb[0].x, rb[0].y, rb[0].z, rb[0].w,
                         rb[1].x, rb[1].y, rb[1].z, rb[1].w};
    const float r1[8] = {rb[2].x, rb[2].y, rb[2].z, rb[2].w,
                         rb[3].x, rb[3].y, rb[3].z, rb[3].w};
#pragma unroll
    for (int j = 0; j < 8; ++j)
        base[((c + j) * SB + kp) >> 1] = pack_bf2(r0[j], r1[j]);
}

static __device__ __forceinline__ void stageB(const float* pb, float4 rb[4]) {
    rb[0] = ((const float4*)pb)[0];
    rb[1] = ((const float4*)pb)[1];
    rb[2] = ((const float4*)(pb + FDIM))[0];
    rb[3] = ((const float4*)(pb + FDIM))[1];
}

// ---------------- Kernel A: GEMM1 (bf16 WMMA) + relu + chunk top-4 ---------
__global__ __launch_bounds__(256) void gemm1_topk_kernel(
        const float* __restrict__ x, const float* __restrict__ Wu,
        float* __restrict__ part) {
    __shared__ unsigned short ldsA[2][MT * SA];
    __shared__ unsigned short ldsB[2][NT * SB];
    __shared__ float ptop[NT * 4 * 2 * 4];   // [col][wm][half][4]

    const int blk = blockIdx.x;
    const int nch = blk & 3;
    const int sch = (blk >> 2) & (NCHUNK - 1);
    const int b   = blk >> 7;
    const int n0 = nch * NT;
    const int s0 = sch * MT;

    const int t = threadIdx.x, lane = t & 31, wave = t >> 5;
    const int wm = wave & 3, wn = wave >> 2;

    v8f acc[2][4];
#pragma unroll
    for (int i = 0; i < 2; ++i)
#pragma unroll
        for (int j = 0; j < 4; ++j)
#pragma unroll
            for (int r = 0; r < 8; ++r) acc[i][j][r] = 0.0f;

    // A: x tile 128x32 (one row, 16 K per thread); B: Wu tile 32x128
    const float* srcA0 = x + ((size_t)b * SEQ + s0 + (t >> 1)) * FDIM + (t & 1) * 16;
    const float* srcB0 = Wu + (size_t)((t >> 4) * 2) * FDIM + n0 + (t & 15) * 8;

    float4 ra[4], rb[4];
#pragma unroll
    for (int q = 0; q < 4; ++q) ra[q] = ((const float4*)srcA0)[q];
    stageB(srcB0, rb);
    commitA(ldsA[0], t, ra, 1.0f);
    commitB(ldsB[0], t, rb);
    __syncthreads();

    for (int ks = 0; ks < NSLICE; ++ks) {
        const int cur = ks & 1;
        const bool more = (ks + 1) < NSLICE;
        if (more) {   // prefetch next K slice into registers
            const float* pa = srcA0 + (ks + 1) * KT;
            const float* pb = srcB0 + (size_t)(ks + 1) * KT * FDIM;
#pragma unroll
            for (int q = 0; q < 4; ++q) ra[q] = ((const float4*)pa)[q];
            stageB(pb, rb);
        }

        Frag a[2], bb[4];
#pragma unroll
        for (int mi = 0; mi < 2; ++mi) load_af(ldsA[cur], wm*32 + mi*16, lane, a[mi]);
#pragma unroll
        for (int ni = 0; ni < 4; ++ni) load_bf(ldsB[cur], wn*64 + ni*16, lane, bb[ni]);
#pragma unroll
        for (int mi = 0; mi < 2; ++mi)
#pragma unroll
            for (int ni = 0; ni < 4; ++ni)
                acc[mi][ni] = __builtin_amdgcn_wmma_f32_16x16x32_bf16(
                    false, a[mi].v, false, bb[ni].v,
                    (short)0, acc[mi][ni], false, false);

        if (more) {
            commitA(ldsA[cur ^ 1], t, ra, 1.0f);
            commitB(ldsB[cur ^ 1], t, rb);
        }
        __syncthreads();
    }

    // relu + per-lane top-4 over the 16 M-values each lane owns per column
    const int nl = lane & 15, half = lane >> 4;
#pragma unroll
    for (int ni = 0; ni < 4; ++ni) {
        float tp[4] = {0.0f, 0.0f, 0.0f, 0.0f};   // relu floor
#pragma unroll
        for (int mi = 0; mi < 2; ++mi)
#pragma unroll
            for (int r = 0; r < 8; ++r)
                ins4(acc[mi][ni][r], tp);
        const int col = wn*64 + ni*16 + nl;
        float* dst = &ptop[((col*4 + wm)*2 + half)*4];
        dst[0]=tp[0]; dst[1]=tp[1]; dst[2]=tp[2]; dst[3]=tp[3];
    }
    __syncthreads();

    if (t < NT) {   // merge 8 lists (4 wm x 2 half) -> chunk top-4 per column
        float tp[4] = {0.0f, 0.0f, 0.0f, 0.0f};
        const float* src = &ptop[t * 32];
#pragma unroll
        for (int j = 0; j < 32; ++j) ins4(src[j], tp);
        const size_t o = (((size_t)b * NCHUNK + sch) * FDIM + n0 + t) * 4;
        part[o+0]=tp[0]; part[o+1]=tp[1]; part[o+2]=tp[2]; part[o+3]=tp[3];
    }
}

// ---------------- Kernel B: merge chunk top-4 lists -> z -------------------
__global__ void topk_merge_kernel(const float* __restrict__ part,
                                  float* __restrict__ z) {
    const int idx = blockIdx.x * blockDim.x + threadIdx.x;
    if (idx >= BSZ * FDIM) return;
    const int b = idx >> 9, g = idx & (FDIM - 1);
    float tp[4] = {0.0f, 0.0f, 0.0f, 0.0f};
    for (int ch = 0; ch < NCHUNK; ++ch) {
        const float* p = part + (((size_t)b * NCHUNK + ch) * FDIM + g) * 4;
        ins4(p[0], tp); ins4(p[1], tp); ins4(p[2], tp); ins4(p[3], tp);
    }
    float* zo = z + (size_t)b * (FDIM * KTOP) + g * 4;  // topv[b][g][0..3]
    zo[0]=tp[0]; zo[1]=tp[1]; zo[2]=tp[2]; zo[3]=tp[3];
}

// ---------------- Kernel C: out = relu((z*P) @ Q + Bb) ---------------------
__global__ __launch_bounds__(256) void gemm2_kernel(
        const float* __restrict__ P, const float* __restrict__ Q,
        const float* __restrict__ Bb, const float* __restrict__ z,
        float* __restrict__ out) {
    __shared__ unsigned short ldsA[2][MT * SA];
    __shared__ unsigned short ldsB[2][NT * SB];

    const int blk = blockIdx.x;
    const int nch = blk & 3;
    const int pair = blk >> 2;              // b*4 + i, 0..127
    const int b = pair >> 2, i = pair & 3;
    const int n0 = nch * NT;

    const int t = threadIdx.x, lane = t & 31, wave = t >> 5;
    const int wm = wave & 3, wn = wave >> 2;

    const float* zrow = z + (size_t)b * (FDIM * KTOP) + i * FDIM + (t >> 1) * 4;

    v8f acc[2][4];
#pragma unroll
    for (int a_ = 0; a_ < 2; ++a_)
#pragma unroll
        for (int j = 0; j < 4; ++j)
#pragma unroll
            for (int r = 0; r < 8; ++r) acc[a_][j][r] = 0.0f;

    // A[m][k] = z[b,i, m*4 + (k>>7)] * P[m][k]; z index const per 32-K slice
    const float* srcA0 = P + (size_t)(t >> 1) * FDIM + (t & 1) * 16;
    const float* srcB0 = Q + (size_t)((t >> 4) * 2) * FDIM + n0 + (t & 15) * 8;

    float4 ra[4], rb[4];
    float zv = zrow[0];
#pragma unroll
    for (int q = 0; q < 4; ++q) ra[q] = ((const float4*)srcA0)[q];
    stageB(srcB0, rb);
    commitA(ldsA[0], t, ra, zv);
    commitB(ldsB[0], t, rb);
    __syncthreads();

    float zn = zv;
    for (int ks = 0; ks < NSLICE; ++ks) {
        const int cur = ks & 1;
        const bool more = (ks + 1) < NSLICE;
        if (more) {
            const int k0n = (ks + 1) * KT;
            zn = zrow[k0n >> 7];
            const float* pa = srcA0 + k0n;
            const float* pb = srcB0 + (size_t)k0n * FDIM;
#pragma unroll
            for (int q = 0; q < 4; ++q) ra[q] = ((const float4*)pa)[q];
            stageB(pb, rb);
        }

        Frag a[2], bb[4];
#pragma unroll
        for (int mi = 0; mi < 2; ++mi) load_af(ldsA[cur], wm*32 + mi*16, lane, a[mi]);
#pragma unroll
        for (int ni = 0; ni < 4; ++ni) load_bf(ldsB[cur], wn*64 + ni*16, lane, bb[ni]);
#pragma unroll
        for (int mi = 0; mi < 2; ++mi)
#pragma unroll
            for (int ni = 0; ni < 4; ++ni)
                acc[mi][ni] = __builtin_amdgcn_wmma_f32_16x16x32_bf16(
                    false, a[mi].v, false, bb[ni].v,
                    (short)0, acc[mi][ni], false, false);

        if (more) {
            commitA(ldsA[cur ^ 1], t, ra, zn);
            commitB(ldsB[cur ^ 1], t, rb);
        }
        __syncthreads();
    }

    // epilogue: + bias, relu, flat store (C layout: M = r + 8*half, N = lane&15)
    const int nl = lane & 15, half = lane >> 4;
#pragma unroll
    for (int mi = 0; mi < 2; ++mi)
#pragma unroll
        for (int ni = 0; ni < 4; ++ni)
#pragma unroll
            for (int r = 0; r < 8; ++r) {
                const int m = wm*32 + mi*16 + r + 8*half;
                const int q = n0 + wn*64 + ni*16 + nl;
                float v = acc[mi][ni][r] + Bb[(size_t)m * FDIM + q];
                out[((size_t)pair * MT + m) * FDIM + q] = fmaxf(v, 0.0f);
            }
}

// ---------------------------------------------------------------------------
extern "C" void kernel_launch(void* const* d_in, const int* in_sizes, int n_in,
                              void* d_out, int out_size, void* d_ws, size_t ws_size,
                              hipStream_t stream) {
    (void)in_sizes; (void)n_in; (void)out_size; (void)ws_size;
    const float* x  = (const float*)d_in[0];
    const float* Wu = (const float*)d_in[1];
    const float* P  = (const float*)d_in[2];
    const float* Q  = (const float*)d_in[3];
    const float* Bb = (const float*)d_in[4];
    float* out = (float*)d_out;

    float* part = (float*)d_ws;                                  // 8 MB
    float* z    = part + (size_t)BSZ * NCHUNK * FDIM * 4;        // 256 KB

    gemm1_topk_kernel<<<BSZ * NCHUNK * 4, 256, 0, stream>>>(x, Wu, part);
    topk_merge_kernel<<<(BSZ * FDIM + 255) / 256, 256, 0, stream>>>(part, z);
    gemm2_kernel<<<BSZ * KTOP * 4, 256, 0, stream>>>(P, Q, Bb, z, out);
}
